// MultiHeadAttentionWithSWA_1125281431947
// MI455X (gfx1250) — matrix operand order repaired and checked by reading, not compiled, not run on hardware
//
#include <hip/hip_runtime.h>

// Problem constants (match reference)
#define BB   4
#define SS   2048
#define DD   1024
#define HH   16
#define WIN  512
#define HD   64

typedef __bf16 bf16;
typedef __attribute__((ext_vector_type(16))) __bf16 v16bf;
typedef __attribute__((ext_vector_type(8)))  float  v8f;
typedef __attribute__((ext_vector_type(4)))  int    v4i;

typedef __attribute__((address_space(1))) v4i* gptr_v4i;
typedef __attribute__((address_space(3))) v4i* lptr_v4i;

union ABFrag { v16bf v; uint4 q[2]; };

__device__ __forceinline__ v8f wmma_bf16(v16bf a, v16bf b, v8f c) {
  return __builtin_amdgcn_wmma_f32_16x16x32_bf16(false, a, false, b, (short)0, c, false, false);
}

#if __has_builtin(__builtin_amdgcn_global_load_async_to_lds_b128)
#define ASYNC_CP 1
#else
#define ASYNC_CP 0
#endif

__device__ __forceinline__ void wait_async0() {
#if __has_builtin(__builtin_amdgcn_s_wait_asynccnt)
  __builtin_amdgcn_s_wait_asynccnt(0);
#else
  asm volatile("s_wait_asynccnt 0x0" ::: "memory");
#endif
}

// ---------------------------------------------------------------------------
// Kernel 1: convert x (f32) -> bf16, row-major [B*S, D]. 8 elems/thread.
// ---------------------------------------------------------------------------
__global__ __launch_bounds__(256) void cvt_x_kernel(const float* __restrict__ x,
                                                    bf16* __restrict__ xb) {
  long t = ((long)blockIdx.x * 256 + threadIdx.x) * 8;
  float4 f0 = *(const float4*)(x + t);
  float4 f1 = *(const float4*)(x + t + 4);
  union { bf16 h[8]; uint4 u; } pk;
  pk.h[0] = (bf16)f0.x; pk.h[1] = (bf16)f0.y;
  pk.h[2] = (bf16)f0.z; pk.h[3] = (bf16)f0.w;
  pk.h[4] = (bf16)f1.x; pk.h[5] = (bf16)f1.y;
  pk.h[6] = (bf16)f1.z; pk.h[7] = (bf16)f1.w;
  *(uint4*)(xb + t) = pk.u;
}

// ---------------------------------------------------------------------------
// Kernel 2: pack Wq,Wk,Wv,Wo (f32 [D,D], k-major) into bf16 WMMA B-fragments.
//   idx = ((w*NT + nt)*KT + kt)*512 + lane*16 + e
//   element e of lane l := W[kt*32 + (e>>3)*16 + (l>>4)*8 + (e&7)][nt*16 + (l&15)]
// ---------------------------------------------------------------------------
__global__ __launch_bounds__(256) void pack_w_kernel(const float* __restrict__ Wq,
                                                     const float* __restrict__ Wk,
                                                     const float* __restrict__ Wv,
                                                     const float* __restrict__ Wo,
                                                     bf16* __restrict__ wp) {
  const int KT = DD / 32, NT = DD / 16;           // 32, 64
  long t = (long)blockIdx.x * 256 + threadIdx.x;  // 4*D*D threads
  int e    = (int)(t & 15);
  int lane = (int)((t >> 4) & 31);
  long frag = t >> 9;
  int kt = (int)(frag % KT);
  int nt = (int)((frag / KT) % NT);
  int w  = (int)(frag / ((long)KT * NT));
  int k = kt * 32 + ((e >> 3) << 4) + ((lane >> 4) << 3) + (e & 7);
  int n = nt * 16 + (lane & 15);
  const float* W = (w == 0) ? Wq : (w == 1) ? Wk : (w == 2) ? Wv : Wo;
  wp[t] = (bf16)W[(long)k * DD + n];
}

// ---------------------------------------------------------------------------
// Kernel 3: fused QKV projection with LDS-staged B (async copy, double
// buffered, one barrier per K-step). Block tile 128x128 (8 waves, 4x2);
// wave tile 32x64. A direct from global, register ping-pong (no copies).
// Q,K stored bf16 [B,H,S,HD]; V stored transposed bf16 [B,H,HD,S].
// ---------------------------------------------------------------------------
__global__ __launch_bounds__(256) void qkv_gemm_kernel(const bf16* __restrict__ xb,
                                                       const bf16* __restrict__ wp,
                                                       bf16* __restrict__ Qb,
                                                       bf16* __restrict__ Kb,
                                                       bf16* __restrict__ VT) {
  const int KT = DD / 32, NT = DD / 16;
  __shared__ bf16 smem[2][8][512];      // 2 buffers x 8 frags x 1KB = 16KB
  int w    = blockIdx.z;                // 0=Q 1=K 2=V
  int wave = threadIdx.x >> 5;
  int lane = threadIdx.x & 31;
  int l15  = lane & 15;
  int hl   = lane >> 4;
  int wr   = wave >> 1;                 // 0..3
  int wc   = wave & 1;                  // 0..1
  int m0   = blockIdx.x * 128 + wr * 32;
  int n0   = blockIdx.y * 128 + wc * 64;
  int fbase = wc * 4;
  long wbase = (long)w * NT + blockIdx.y * 8;   // first frag row of this block

  v8f acc[2][4] = {};
  const bf16* arow0 = xb + (long)(m0 + l15) * DD + hl * 8;
  const bf16* arow1 = arow0 + 16 * DD;

  // Cooperative stage of one 8KB B k-slice into LDS buffer `buf`.
  int ct = threadIdx.x;
  int cf = ct >> 5, cl = ct & 31;       // frag 0..7, 32B slice 0..31
  auto stageB = [&](int buf, int kt) {
    const bf16* src = wp + (((wbase + cf) * KT + kt) << 9) + cl * 16;
#if ASYNC_CP
    __builtin_amdgcn_global_load_async_to_lds_b128(
        (gptr_v4i)(src), (lptr_v4i)(&smem[buf][cf][cl * 16]), 0, 0);
    __builtin_amdgcn_global_load_async_to_lds_b128(
        (gptr_v4i)(src + 8), (lptr_v4i)(&smem[buf][cf][cl * 16 + 8]), 0, 0);
#else
    uint4 t0 = *(const uint4*)(src);
    uint4 t1 = *(const uint4*)(src + 8);
    *(uint4*)&smem[buf][cf][cl * 16]     = t0;
    *(uint4*)&smem[buf][cf][cl * 16 + 8] = t1;
#endif
  };

  ABFrag aC[2], aN[2];
  auto LDA = [&](ABFrag* f, int kt) {
    f[0].q[0] = *(const uint4*)(arow0 + kt * 32);
    f[0].q[1] = *(const uint4*)(arow0 + kt * 32 + 16);
    f[1].q[0] = *(const uint4*)(arow1 + kt * 32);
    f[1].q[1] = *(const uint4*)(arow1 + kt * 32 + 16);
  };
  auto STEP = [&](ABFrag* a, int buf) {
#pragma unroll
    for (int nt = 0; nt < 4; ++nt) {
      ABFrag bf_;
      const bf16* lp = &smem[buf][fbase + nt][lane * 16];
      bf_.q[0] = *(const uint4*)(lp);
      bf_.q[1] = *(const uint4*)(lp + 8);
      acc[0][nt] = wmma_bf16(a[0].v, bf_.v, acc[0][nt]);
      acc[1][nt] = wmma_bf16(a[1].v, bf_.v, acc[1][nt]);
    }
  };

  // Prologue: stage k-slice 0, load A slice 0.
  stageB(0, 0);
  LDA(aC, 0);
  wait_async0();
  __syncthreads();

  for (int kt = 0; kt < KT; kt += 2) {
    // even step: compute from buf0 while staging kt+1 into buf1
    stageB(1, kt + 1);
    LDA(aN, kt + 1);
    STEP(aC, 0);
    wait_async0();
    __syncthreads();
    // odd step: compute from buf1 while staging kt+2 into buf0
    if (kt + 2 < KT) { stageB(0, kt + 2); LDA(aC, kt + 2); }
    STEP(aN, 1);
    wait_async0();
    __syncthreads();
  }

  // Epilogue. C/D layout: M = r + 8*hl, N = lane&15.
#pragma unroll
  for (int mi = 0; mi < 2; ++mi) {
    int m  = m0 + mi * 16;
    int bb = m >> 11;                  // m / S
    int sb = (m & (SS - 1)) + 8 * hl;  // s for r=0
    if (w < 2) {
      bf16* base = (w == 0 ? Qb : Kb) + (long)bb * HH * SS * HD;
#pragma unroll
      for (int nt = 0; nt < 4; ++nt) {
        int n = n0 + nt * 16 + l15;
        int hh = n >> 6, dd = n & 63;
        bf16* p = base + ((long)hh * SS + sb) * HD + dd;
#pragma unroll
        for (int r = 0; r < 8; ++r) p[(long)r * HD] = (bf16)acc[mi][nt][r];
      }
    } else {
#pragma unroll
      for (int nt = 0; nt < 4; ++nt) {
        int n = n0 + nt * 16 + l15;
        int hh = n >> 6, dd = n & 63;
        union { bf16 h[8]; uint4 u; } pk;
#pragma unroll
        for (int r = 0; r < 8; ++r) pk.h[r] = (bf16)acc[mi][nt][r];
        *(uint4*)(VT + (((long)(bb * HH + hh) * HD + dd) * SS + sb)) = pk.u;
      }
    }
  }
}

// ---------------------------------------------------------------------------
// Kernel 4: sliding-window flash attention, pipelined (register double
// buffering of K frags; V frags + next-step K issued before score WMMAs).
// ---------------------------------------------------------------------------
__global__ __launch_bounds__(256) void attn_kernel(const bf16* __restrict__ Qb,
                                                   const bf16* __restrict__ Kb,
                                                   const bf16* __restrict__ VT,
                                                   bf16* __restrict__ ctx) {
  __shared__ bf16 plds[8][16][40];     // per-wave 16x32 P tile (padded)
  int wave = threadIdx.x >> 5;
  int lane = threadIdx.x & 31;
  int l15  = lane & 15;
  int hl   = lane >> 4;

  int gw = blockIdx.x * 8 + wave;      // 0..8191
  int qt = gw & (SS / 16 - 1);
  int h  = (gw >> 7) & (HH - 1);
  int b  = gw >> 11;
  int q0 = qt * 16;

  const bf16* Qh = Qb + (long)(b * HH + h) * SS * HD;
  const bf16* Kh = Kb + (long)(b * HH + h) * SS * HD;
  const bf16* Vh = VT + (long)(b * HH + h) * HD * SS;

  ABFrag aq0, aq1;
  const bf16* qrow = Qh + (long)(q0 + l15) * HD;
  aq0.q[0] = *(const uint4*)(qrow + hl * 16);
  aq0.q[1] = *(const uint4*)(qrow + hl * 16 + 8);
  aq1.q[0] = *(const uint4*)(qrow + 32 + hl * 16);
  aq1.q[1] = *(const uint4*)(qrow + 32 + hl * 16 + 8);

  float m[8], l[8];
  v8f oacc[4] = {};
#pragma unroll
  for (int r = 0; r < 8; ++r) { m[r] = -1e30f; l[r] = 0.f; }

  int jlo = q0 + 15 - (WIN - 1);
  if (jlo < 0) jlo = 0;
  jlo &= ~31;
  int jhi = q0 + 15;
  const float sc = 0.125f;             // 1/sqrt(HD)

  ABFrag bk[4], bkn[4], bv[4];
  auto LDK = [&](ABFrag* f, int j) {
    const bf16* kp = Kh + (long)(j + l15) * HD + hl * 16;
    f[0].q[0] = *(const uint4*)(kp);       f[0].q[1] = *(const uint4*)(kp + 8);
    f[1].q[0] = *(const uint4*)(kp + 32);  f[1].q[1] = *(const uint4*)(kp + 40);
    const bf16* kp2 = kp + 16 * HD;
    f[2].q[0] = *(const uint4*)(kp2);      f[2].q[1] = *(const uint4*)(kp2 + 8);
    f[3].q[0] = *(const uint4*)(kp2 + 32); f[3].q[1] = *(const uint4*)(kp2 + 40);
  };

  LDK(bk, jlo);
  for (int j0 = jlo; j0 <= jhi; j0 += 32) {
#pragma unroll
    for (int nt = 0; nt < 4; ++nt) {
      const bf16* vp = Vh + (long)(nt * 16 + l15) * SS + j0 + hl * 16;
      bv[nt].q[0] = *(const uint4*)(vp);
      bv[nt].q[1] = *(const uint4*)(vp + 8);
    }
    bool hasNext = (j0 + 32 <= jhi);
    if (hasNext) LDK(bkn, j0 + 32);

    v8f s0 = {}, s1 = {};
    s0 = wmma_bf16(aq0.v, bk[0].v, s0);
    s0 = wmma_bf16(aq1.v, bk[1].v, s0);
    s1 = wmma_bf16(aq0.v, bk[2].v, s1);
    s1 = wmma_bf16(aq1.v, bk[3].v, s1);

#pragma unroll
    for (int r = 0; r < 8; ++r) {
      int i  = q0 + r + 8 * hl;
      int d1 = i - (j0 + l15);
      int d2 = i - (j0 + 16 + l15);
      s0[r] = (d1 >= 0 && d1 < WIN) ? s0[r] * sc : -1e30f;
      s1[r] = (d2 >= 0 && d2 < WIN) ? s1[r] * sc : -1e30f;
    }

#pragma unroll
    for (int r = 0; r < 8; ++r) {
      float t = fmaxf(s0[r], s1[r]);
      t = fmaxf(t, __shfl_xor(t, 1, 32));
      t = fmaxf(t, __shfl_xor(t, 2, 32));
      t = fmaxf(t, __shfl_xor(t, 4, 32));
      t = fmaxf(t, __shfl_xor(t, 8, 32));
      float mn = fmaxf(m[r], t);
      float alpha = __expf(m[r] - mn);
      m[r] = mn;
      float p0 = __expf(s0[r] - mn);
      float p1 = __expf(s1[r] - mn);
      float rs = p0 + p1;
      rs += __shfl_xor(rs, 1, 32);
      rs += __shfl_xor(rs, 2, 32);
      rs += __shfl_xor(rs, 4, 32);
      rs += __shfl_xor(rs, 8, 32);
      l[r] = l[r] * alpha + rs;
#pragma unroll
      for (int nt = 0; nt < 4; ++nt) oacc[nt][r] *= alpha;
      plds[wave][r + 8 * hl][l15]      = (bf16)p0;
      plds[wave][r + 8 * hl][16 + l15] = (bf16)p1;
    }

    // LDS is in-order per wave; wait + compiler barrier, then read P as A-frag.
    asm volatile("s_wait_dscnt 0" ::: "memory");
    ABFrag ap;
    const bf16* prow = &plds[wave][l15][0];
    ap.q[0] = *(const uint4*)(prow + hl * 16);
    ap.q[1] = *(const uint4*)(prow + hl * 16 + 8);

#pragma unroll
    for (int nt = 0; nt < 4; ++nt)
      oacc[nt] = wmma_bf16(ap.v, bv[nt].v, oacc[nt]);

    if (hasNext) {
#pragma unroll
      for (int nt = 0; nt < 4; ++nt) bk[nt] = bkn[nt];
    }
  }

  bf16* cbase = ctx + ((long)(b * SS + q0)) * DD + h * 64;
#pragma unroll
  for (int r = 0; r < 8; ++r) {
    float inv = 1.f / l[r];
#pragma unroll
    for (int nt = 0; nt < 4; ++nt)
      cbase[(long)(r + 8 * hl) * DD + nt * 16 + l15] = (bf16)(oacc[nt][r] * inv);
  }
}

// ---------------------------------------------------------------------------
// Kernel 5: output projection. out = ctx @ Wo + bo, f32 to d_out.
// Same LDS-staged, async double-buffered structure as the QKV GEMM.
// ---------------------------------------------------------------------------
__global__ __launch_bounds__(256) void out_gemm_kernel(const bf16* __restrict__ ctx,
                                                       const bf16* __restrict__ wp,
                                                       const float* __restrict__ bo,
                                                       float* __restrict__ out) {
  const int KT = DD / 32, NT = DD / 16;
  __shared__ bf16 smem[2][8][512];
  const int w = 3;                     // Wo slot
  int wave = threadIdx.x >> 5;
  int lane = threadIdx.x & 31;
  int l15  = lane & 15;
  int hl   = lane >> 4;
  int wr   = wave >> 1;
  int wc   = wave & 1;
  int m0   = blockIdx.x * 128 + wr * 32;
  int n0   = blockIdx.y * 128 + wc * 64;
  int fbase = wc * 4;
  long wbase = (long)w * NT + blockIdx.y * 8;

  v8f acc[2][4] = {};
  const bf16* arow0 = ctx + (long)(m0 + l15) * DD + hl * 8;
  const bf16* arow1 = arow0 + 16 * DD;

  int ct = threadIdx.x;
  int cf = ct >> 5, cl = ct & 31;
  auto stageB = [&](int buf, int kt) {
    const bf16* src = wp + (((wbase + cf) * KT + kt) << 9) + cl * 16;
#if ASYNC_CP
    __builtin_amdgcn_global_load_async_to_lds_b128(
        (gptr_v4i)(src), (lptr_v4i)(&smem[buf][cf][cl * 16]), 0, 0);
    __builtin_amdgcn_global_load_async_to_lds_b128(
        (gptr_v4i)(src + 8), (lptr_v4i)(&smem[buf][cf][cl * 16 + 8]), 0, 0);
#else
    uint4 t0 = *(const uint4*)(src);
    uint4 t1 = *(const uint4*)(src + 8);
    *(uint4*)&smem[buf][cf][cl * 16]     = t0;
    *(uint4*)&smem[buf][cf][cl * 16 + 8] = t1;
#endif
  };

  ABFrag aC[2], aN[2];
  auto LDA = [&](ABFrag* f, int kt) {
    f[0].q[0] = *(const uint4*)(arow0 + kt * 32);
    f[0].q[1] = *(const uint4*)(arow0 + kt * 32 + 16);
    f[1].q[0] = *(const uint4*)(arow1 + kt * 32);
    f[1].q[1] = *(const uint4*)(arow1 + kt * 32 + 16);
  };
  auto STEP = [&](ABFrag* a, int buf) {
#pragma unroll
    for (int nt = 0; nt < 4; ++nt) {
      ABFrag bf_;
      const bf16* lp = &smem[buf][fbase + nt][lane * 16];
      bf_.q[0] = *(const uint4*)(lp);
      bf_.q[1] = *(const uint4*)(lp + 8);
      acc[0][nt] = wmma_bf16(a[0].v, bf_.v, acc[0][nt]);
      acc[1][nt] = wmma_bf16(a[1].v, bf_.v, acc[1][nt]);
    }
  };

  stageB(0, 0);
  LDA(aC, 0);
  wait_async0();
  __syncthreads();

  for (int kt = 0; kt < KT; kt += 2) {
    stageB(1, kt + 1);
    LDA(aN, kt + 1);
    STEP(aC, 0);
    wait_async0();
    __syncthreads();
    if (kt + 2 < KT) { stageB(0, kt + 2); LDA(aC, kt + 2); }
    STEP(aN, 1);
    wait_async0();
    __syncthreads();
  }

#pragma unroll
  for (int mi = 0; mi < 2; ++mi) {
    int m = m0 + mi * 16;
#pragma unroll
    for (int nt = 0; nt < 4; ++nt) {
      int n = n0 + nt * 16 + l15;
      float bias = bo[n];
      float* p = out + (long)(m + 8 * hl) * DD + n;
#pragma unroll
      for (int r = 0; r < 8; ++r) p[(long)r * DD] = acc[mi][nt][r] + bias;
    }
  }
}

// ---------------------------------------------------------------------------
// Host launcher
// ---------------------------------------------------------------------------
extern "C" void kernel_launch(void* const* d_in, const int* in_sizes, int n_in,
                              void* d_out, int out_size, void* d_ws, size_t ws_size,
                              hipStream_t stream) {
  const float* x  = (const float*)d_in[0];
  const float* Wq = (const float*)d_in[1];
  const float* Wk = (const float*)d_in[2];
  const float* Wv = (const float*)d_in[3];
  const float* Wo = (const float*)d_in[4];
  const float* bo = (const float*)d_in[5];
  float* out = (float*)d_out;

  char* ws = (char*)d_ws;
  bf16* xb = (bf16*)ws;                            // 16 MiB; reused as ctx
  bf16* wp = (bf16*)(ws + (16ull << 20));          //  8 MiB packed weights
  bf16* Qb = (bf16*)(ws + (24ull << 20));          // 16 MiB [B,H,S,HD]
  bf16* Kb = (bf16*)(ws + (40ull << 20));          // 16 MiB [B,H,S,HD]
  bf16* VT = (bf16*)(ws + (56ull << 20));          // 16 MiB [B,H,HD,S]
  bf16* ctx = xb;                                  // alias: xb dead after QKV

  cvt_x_kernel<<<4096, 256, 0, stream>>>(x, xb);
  pack_w_kernel<<<16384, 256, 0, stream>>>(Wq, Wk, Wv, Wo, wp);
  qkv_gemm_kernel<<<dim3(64, 8, 3), 256, 0, stream>>>(xb, wp, Qb, Kb, VT);
  attn_kernel<<<1024, 256, 0, stream>>>(Qb, Kb, VT, ctx);
  out_gemm_kernel<<<dim3(64, 8), 256, 0, stream>>>(ctx, wp, bo, out);
}
